// QCMatrixMultiHeadAttention_8641474200369
// MI455X (gfx1250) — matrix operand order repaired and checked by reading, not compiled, not run on hardware
//
#include <hip/hip_runtime.h>
#include <hip/hip_bf16.h>
#include <math.h>

// ---------------------------------------------------------------------------
// QC (quaternion-complex) multi-head attention for MI455X (gfx1250, wave32).
// All GEMMs on v_wmma_f32_16x16x32_bf16. fp32 inputs converted to bf16 once;
// weights converted PRE-TRANSPOSED and K written transposed by its projection,
// but B-fragments are gathered per element; only A-fragments use b128 loads.
// Flash-style online-softmax attention; only LDS use is the P-tile transpose.
// ---------------------------------------------------------------------------

typedef __attribute__((ext_vector_type(16))) __bf16 v16bf;
typedef __attribute__((ext_vector_type(8)))  __bf16 v8bf;
typedef __attribute__((ext_vector_type(8)))  float  v8f;

#define SEQ  2048
#define EMB  1024
#define NH   16
#define HD   64

__device__ __forceinline__ v8f v8f_zero() {
    v8f z = {0.f, 0.f, 0.f, 0.f, 0.f, 0.f, 0.f, 0.f};
    return z;
}

__device__ __forceinline__ v8f wmma_bf16(v16bf a, v16bf b, v8f c) {
    return __builtin_amdgcn_wmma_f32_16x16x32_bf16(false, a, false, b, (short)0, c,
                                                   false, false);
}

__device__ __forceinline__ v16bf combine16(v8bf a, v8bf b) {
    return __builtin_shufflevector(a, b, 0, 1, 2, 3, 4, 5, 6, 7,
                                   8, 9, 10, 11, 12, 13, 14, 15);
}

// A-fragment (bf16 16x32 MxK), ISA 7.12.2: lane l -> m = l&15,
// element e -> k = ((e&8)<<1) + 8*(l>=16) + (e&7): two contiguous 8-element
// runs per lane at k = khi and khi+16 -> two b128 loads.
__device__ __forceinline__ v16bf load_afrag_bf(const __bf16* p, int ld, int lane) {
    const int m   = lane & 15;
    const int khi = (lane >> 4) << 3;
    const __bf16* q = p + m * ld + khi;
    return combine16(*(const v8bf*)q, *(const v8bf*)(q + 16));
}

// B-fragment (bf16 32x16 KxN) from K-major memory: B(k,n) = p[k*ld + n].
// Lane l holds column n=l&15, element e -> k = ((e&8)<<1) + 8*(l>=16) + (e&7).
__device__ __forceinline__ v16bf load_bfrag_row_bf(const __bf16* p, size_t ld,
                                                   int lane) {
    const int n  = lane & 15;
    const int kb = (lane >> 4) << 3;
    v16bf r;
#pragma unroll
    for (int e = 0; e < 16; ++e) {
        const int k = ((e & 8) << 1) + kb + (e & 7);
        r[e] = p[(size_t)k * ld + n];
    }
    return r;
}

// xor-shuffle reductions confined to each 16-lane half (wave32).
__device__ __forceinline__ float redmax16(float v) {
#pragma unroll
    for (int m = 1; m <= 8; m <<= 1) v = fmaxf(v, __shfl_xor(v, m, 32));
    return v;
}
__device__ __forceinline__ float redsum16(float v) {
#pragma unroll
    for (int m = 1; m <= 8; m <<= 1) v += __shfl_xor(v, m, 32);
    return v;
}

// ---------------------------------------------------------------------------
// One-time fp32 -> bf16 conversions.
// ---------------------------------------------------------------------------
__global__ void __launch_bounds__(256) cvt_kernel(const float* __restrict__ src,
                                                  __bf16* __restrict__ dst) {
    const int i = blockIdx.x * 256 + threadIdx.x;
    const float4* s = (const float4*)src;
    float4 a = s[2 * i], b = s[2 * i + 1];
    float v[8] = {a.x, a.y, a.z, a.w, b.x, b.y, b.z, b.w};
    v8bf d;
#pragma unroll
    for (int e = 0; e < 8; ++e) d[e] = (__bf16)v[e];
    ((v8bf*)dst)[i] = d;
}

// fp32 -> {bf16, bf16 * j2} pair (j2 = -1 + sin(2*theta)).
__global__ void __launch_bounds__(256) cvt_pair_kernel(const float* __restrict__ src,
                                                       __bf16* __restrict__ dst,
                                                       __bf16* __restrict__ dstj,
                                                       const float* __restrict__ theta) {
    const float j2 = -1.0f + sinf(2.0f * theta[0]);
    const int i = blockIdx.x * 256 + threadIdx.x;
    const float4* s = (const float4*)src;
    float4 a = s[2 * i], b = s[2 * i + 1];
    float v[8] = {a.x, a.y, a.z, a.w, b.x, b.y, b.z, b.w};
    v8bf d, dj;
#pragma unroll
    for (int e = 0; e < 8; ++e) {
        d[e]  = (__bf16)v[e];
        dj[e] = (__bf16)(v[e] * j2);
    }
    ((v8bf*)dst)[i]  = d;
    ((v8bf*)dstj)[i] = dj;
}

// fp32 W[1024][1024] -> bf16 W^T[1024][1024] via LDS-tiled transpose.
__global__ void __launch_bounds__(256) cvt_transpose_kernel(
    const float* __restrict__ src, __bf16* __restrict__ dst) {
    __shared__ __bf16 t[64][72];           // padded against bank conflicts
    const int r0 = (blockIdx.x & 15) << 6; // source row (n) tile
    const int c0 = (blockIdx.x >> 4) << 6; // source col (k) tile
#pragma unroll
    for (int i = 0; i < 16; ++i) {
        const int lin = threadIdx.x + 256 * i;   // 0..4095
        const int rr = lin >> 6, cc = lin & 63;
        t[rr][cc] = (__bf16)src[(size_t)(r0 + rr) * EMB + c0 + cc];
    }
    __syncthreads();
#pragma unroll
    for (int i = 0; i < 16; ++i) {
        const int lin = threadIdx.x + 256 * i;
        const int kk = lin >> 6, nn = lin & 63;
        dst[(size_t)(c0 + kk) * EMB + r0 + nn] = t[nn][kk];
    }
}

// ---------------------------------------------------------------------------
// QC linear (bf16):  oa = xa@wa.T + ba + xbj@wb.T ; ob = xa@wb.T + bb + xb@wa.T
// Weights arrive pre-transposed (wT[k][n]) and B-fragments are per-element
// loads. One wave owns a 16x64 tile; no LDS, no barriers.
// MODE: 0 = bf16 row-major out, 1 = bf16 TRANSPOSED out ([b, col, s], packed
// b128 stores), 2 = f32 row-major out.
// ---------------------------------------------------------------------------
template <int MODE>
__global__ void __launch_bounds__(128) qc_proj_bf16(
    const __bf16* __restrict__ xa, const __bf16* __restrict__ xb,
    const __bf16* __restrict__ xbj,
    const __bf16* __restrict__ wat, const __bf16* __restrict__ wbt,
    const float* __restrict__ ba, const float* __restrict__ bb,
    float* __restrict__ oaf, float* __restrict__ obf,
    __bf16* __restrict__ oah, __bf16* __restrict__ obh) {
    const int K = EMB, N = EMB;
    const int tid  = threadIdx.x;
    const int lane = tid & 31;
    const int wid  = tid >> 5;
    const int tn   = blockIdx.x & 15;      // 16 column panels of 64
    const int tmb  = blockIdx.x >> 4;      // 64 row panels of 64
    const int m0   = tmb * 64 + wid * 16;
    const int n0   = tn * 64;

    v8f acc_a[4], acc_b[4];
#pragma unroll
    for (int i = 0; i < 4; ++i) { acc_a[i] = v8f_zero(); acc_b[i] = v8f_zero(); }

#pragma unroll 1
    for (int k0 = 0; k0 < K; k0 += 32) {
        v16bf fxa  = load_afrag_bf(xa  + (size_t)m0 * K + k0, K, lane);
        v16bf fxb  = load_afrag_bf(xb  + (size_t)m0 * K + k0, K, lane);
        v16bf fxbj = load_afrag_bf(xbj + (size_t)m0 * K + k0, K, lane);
        __builtin_prefetch(wat + (size_t)(k0 + 32) * N + n0, 0, 1);
        __builtin_prefetch(xa + (size_t)m0 * K + k0 + 32, 0, 1);
#pragma unroll
        for (int nt = 0; nt < 4; ++nt) {
            v16bf fwa = load_bfrag_row_bf(wat + (size_t)k0 * N + n0 + nt * 16, N, lane);
            v16bf fwb = load_bfrag_row_bf(wbt + (size_t)k0 * N + n0 + nt * 16, N, lane);
            acc_a[nt] = wmma_bf16(fxa,  fwa, acc_a[nt]);
            acc_a[nt] = wmma_bf16(fxbj, fwb, acc_a[nt]);
            acc_b[nt] = wmma_bf16(fxa,  fwb, acc_b[nt]);
            acc_b[nt] = wmma_bf16(fxb,  fwa, acc_b[nt]);
        }
    }

    const int mrow = (lane >> 4) << 3;
    const int ncol = lane & 15;
#pragma unroll
    for (int nt = 0; nt < 4; ++nt) {
        const int n = n0 + nt * 16 + ncol;
        const float bva = ba[n];
        const float bvb = bb[n];
        if constexpr (MODE == 1) {
            // Transposed bf16 out: lane holds 8 consecutive s-rows of column n
            // -> one packed b128 store per component.
            const int bb_ = m0 >> 11;                 // batch
            const int s0  = (m0 & (SEQ - 1)) + mrow;  // seq offset, mult of 8
            v8bf pa, pbv;
#pragma unroll
            for (int r = 0; r < 8; ++r) {
                pa[r]  = (__bf16)(acc_a[nt][r] + bva);
                pbv[r] = (__bf16)(acc_b[nt][r] + bvb);
            }
            const size_t o = ((size_t)bb_ * EMB + n) * SEQ + s0;
            *(v8bf*)(oah + o) = pa;
            *(v8bf*)(obh + o) = pbv;
        } else {
#pragma unroll
            for (int r = 0; r < 8; ++r) {
                const size_t o = (size_t)(m0 + mrow + r) * N + n;
                if constexpr (MODE == 2) {
                    oaf[o] = acc_a[nt][r] + bva;
                    obf[o] = acc_b[nt][r] + bvb;
                } else {
                    oah[o] = (__bf16)(acc_a[nt][r] + bva);
                    obh[o] = (__bf16)(acc_b[nt][r] + bvb);
                }
            }
        }
    }
}

// ---------------------------------------------------------------------------
// Flash-style QC attention, bf16 operands, K pre-transposed (kT[b, d, s]).
// One wave per (b, h, 16 query rows). No cross-wave LDS; the only LDS use is
// the per-wave P-tile C-layout -> A-layout transpose.
// ---------------------------------------------------------------------------
__global__ void __launch_bounds__(128) qc_attn_bf16(
    const __bf16* __restrict__ qa_, const __bf16* __restrict__ qb_,
    const __bf16* __restrict__ kat, const __bf16* __restrict__ kbt,
    const __bf16* __restrict__ va_, const __bf16* __restrict__ vb_,
    const unsigned char* __restrict__ mask,
    const float* __restrict__ head_thetas, const float* __restrict__ theta,
    __bf16* __restrict__ ta, __bf16* __restrict__ tb,
    __bf16* __restrict__ tbj) {
    __shared__ __bf16 pbuf[4][16][32];   // per-wave P tile (4 KB)

    const int tid  = threadIdx.x;
    const int lane = tid & 31;
    const int wid  = tid >> 5;
    const int wave = blockIdx.x * 4 + wid;
    const int qt = wave & (SEQ / 16 - 1);
    const int h  = (wave >> 7) & (NH - 1);
    const int b  = wave >> 11;
    const int q0 = qt * 16;

    const float hj2 = -1.0f + sinf(2.0f * head_thetas[h]);
    const float j2  = -1.0f + sinf(2.0f * theta[0]);
    const size_t base  = (size_t)b * SEQ * EMB + h * HD;        // row-major q/v
    const size_t baseT = ((size_t)b * EMB + h * HD) * SEQ;      // transposed K

    v16bf fqa[2], fqb[2], fqbj[2];
#pragma unroll
    for (int t = 0; t < 2; ++t) {
        fqa[t] = load_afrag_bf(qa_ + base + (size_t)q0 * EMB + t * 32, EMB, lane);
        fqb[t] = load_afrag_bf(qb_ + base + (size_t)q0 * EMB + t * 32, EMB, lane);
#pragma unroll
        for (int e = 0; e < 16; ++e)
            fqbj[t][e] = (__bf16)((float)fqb[t][e] * hj2);
    }

    v8f acc_a[4], acc_b[4];
#pragma unroll
    for (int i = 0; i < 4; ++i) { acc_a[i] = v8f_zero(); acc_b[i] = v8f_zero(); }
    float mrun[8], lrun[8];
#pragma unroll
    for (int r = 0; r < 8; ++r) { mrun[r] = -3.0e38f; lrun[r] = 0.0f; }

    const int mrow = (lane >> 4) << 3;
    const int ncol = lane & 15;
    const unsigned char* mbase = mask + (size_t)b * SEQ * SEQ;
    __bf16* pb = &pbuf[wid][0][0];

#pragma unroll 1
    for (int k0 = 0; k0 < SEQ; k0 += 32) {
        __builtin_prefetch(kat + baseT + k0 + 32, 0, 1);
        float tmag[2][8];
#pragma unroll
        for (int t = 0; t < 2; ++t) {
            v8f sa = v8f_zero();
            v8f sb = v8f_zero();
#pragma unroll
            for (int ds = 0; ds < 2; ++ds) {
                // B(d, key) = kT[(d)*S + key] -> lane = key column, d strided by S.
                const size_t ko = baseT + (size_t)(ds * 32) * SEQ + k0 + t * 16;
                v16bf fka = load_bfrag_row_bf(kat + ko, SEQ, lane);
                v16bf fkb = load_bfrag_row_bf(kbt + ko, SEQ, lane);
                sa = wmma_bf16(fqa[ds],  fka, sa);
                sa = wmma_bf16(fqbj[ds], fkb, sa);
                sb = wmma_bf16(fqa[ds],  fkb, sb);
                sb = wmma_bf16(fqb[ds],  fka, sb);
            }
#pragma unroll
            for (int r = 0; r < 8; ++r) {
                float mg = sqrtf(sa[r] * sa[r] + sb[r] * sb[r] + 1e-8f) * 0.125f;
                const int qrow = q0 + mrow + r;
                const int kcol = k0 + t * 16 + ncol;
                if (!mbase[(size_t)qrow * SEQ + kcol]) mg = -1e9f;
                tmag[t][r] = mg;
            }
        }

        float newmax[8], alpha[8];
#pragma unroll
        for (int r = 0; r < 8; ++r) {
            float v = redmax16(fmaxf(tmag[0][r], tmag[1][r]));
            newmax[r] = fmaxf(mrun[r], v);
            alpha[r]  = expf(mrun[r] - newmax[r]);
            mrun[r]   = newmax[r];
        }

        float psum[8];
#pragma unroll
        for (int r = 0; r < 8; ++r) psum[r] = 0.0f;
#pragma unroll
        for (int t = 0; t < 2; ++t)
#pragma unroll
            for (int r = 0; r < 8; ++r) {
                float p = expf(tmag[t][r] - newmax[r]);
                psum[r] += p;
                pb[(mrow + r) * 32 + t * 16 + ncol] = (__bf16)p;   // ds_store
            }
#pragma unroll
        for (int r = 0; r < 8; ++r)
            lrun[r] = lrun[r] * alpha[r] + redsum16(psum[r]);

#pragma unroll
        for (int dt = 0; dt < 4; ++dt)
#pragma unroll
            for (int r = 0; r < 8; ++r) {
                acc_a[dt][r] *= alpha[r];
                acc_b[dt][r] *= alpha[r];
            }

        // All lanes' P stores must land before the cross-lane b128 re-read.
        asm volatile("s_wait_dscnt 0" ::: "memory");
        v16bf fp = load_afrag_bf(pb, 32, lane);          // 2x ds_load_b128

#pragma unroll
        for (int dt = 0; dt < 4; ++dt) {
            // B(key, d) = v[(k0+key)*E + dt*16 + d] (lane = d column, keys strided by E).
            v16bf fva = load_bfrag_row_bf(va_ + base + (size_t)k0 * EMB + dt * 16,
                                          EMB, lane);
            v16bf fvb = load_bfrag_row_bf(vb_ + base + (size_t)k0 * EMB + dt * 16,
                                          EMB, lane);
            acc_a[dt] = wmma_bf16(fp, fva, acc_a[dt]);
            acc_b[dt] = wmma_bf16(fp, fvb, acc_b[dt]);
        }
    }

    float inv[8];
#pragma unroll
    for (int r = 0; r < 8; ++r) inv[r] = (lrun[r] > 0.0f) ? (1.0f / lrun[r]) : 0.0f;

#pragma unroll
    for (int dt = 0; dt < 4; ++dt)
#pragma unroll
        for (int r = 0; r < 8; ++r) {
            const int row = q0 + mrow + r;
            const int col = h * HD + dt * 16 + ncol;     // merged [B,S,E]
            const float av = acc_a[dt][r] * inv[r];
            const float bv = acc_b[dt][r] * inv[r];
            const size_t o = (size_t)(b * SEQ + row) * EMB + col;
            ta[o]  = (__bf16)av;
            tb[o]  = (__bf16)bv;
            tbj[o] = (__bf16)(bv * j2);
        }
}

// ---------------------------------------------------------------------------
// Host launch.
// ---------------------------------------------------------------------------
extern "C" void kernel_launch(void* const* d_in, const int* in_sizes, int n_in,
                              void* d_out, int out_size, void* d_ws, size_t ws_size,
                              hipStream_t stream) {
    (void)in_sizes; (void)n_in; (void)out_size; (void)ws_size;
    const size_t MN = (size_t)2 * SEQ * EMB;   // 4,194,304
    const size_t WN = (size_t)EMB * EMB;       // 1,048,576

    const float* x_q_a = (const float*)d_in[0];
    const float* x_q_b = (const float*)d_in[1];
    const float* x_k_a = (const float*)d_in[2];
    const float* x_k_b = (const float*)d_in[3];
    const float* x_v_a = (const float*)d_in[4];
    const float* x_v_b = (const float*)d_in[5];
    const unsigned char* mask = (const unsigned char*)d_in[6];
    const float* layer_theta  = (const float*)d_in[7];
    const float* q_wa = (const float*)d_in[8];
    const float* q_wb = (const float*)d_in[9];
    const float* q_ba = (const float*)d_in[10];
    const float* q_bb = (const float*)d_in[11];
    const float* k_wa = (const float*)d_in[12];
    const float* k_wb = (const float*)d_in[13];
    const float* k_ba = (const float*)d_in[14];
    const float* k_bb = (const float*)d_in[15];
    const float* v_wa = (const float*)d_in[16];
    const float* v_wb = (const float*)d_in[17];
    const float* v_ba = (const float*)d_in[18];
    const float* v_bb = (const float*)d_in[19];
    const float* o_wa = (const float*)d_in[20];
    const float* o_wb = (const float*)d_in[21];
    const float* o_ba = (const float*)d_in[22];
    const float* o_bb = (const float*)d_in[23];
    const float* head_thetas = (const float*)d_in[24];

    __bf16* w = (__bf16*)d_ws;
    __bf16* xqa  = w;            __bf16* xqb  = w + 1 * MN;  __bf16* xqbj = w + 2 * MN;
    __bf16* xka  = w + 3 * MN;   __bf16* xkb  = w + 4 * MN;  __bf16* xkbj = w + 5 * MN;
    __bf16* xva  = w + 6 * MN;   __bf16* xvb  = w + 7 * MN;  __bf16* xvbj = w + 8 * MN;
    __bf16* wbase = w + 9 * MN;
    __bf16* qwa = wbase + 0 * WN; __bf16* qwb = wbase + 1 * WN;
    __bf16* kwa = wbase + 2 * WN; __bf16* kwb = wbase + 3 * WN;
    __bf16* vwa = wbase + 4 * WN; __bf16* vwb = wbase + 5 * WN;
    __bf16* owa = wbase + 6 * WN; __bf16* owb = wbase + 7 * WN;
    __bf16* pbase = wbase + 8 * WN;
    __bf16* qa  = pbase + 0 * MN; __bf16* qb  = pbase + 1 * MN;
    __bf16* kaT = pbase + 2 * MN; __bf16* kbT = pbase + 3 * MN;  // transposed K
    __bf16* va  = pbase + 4 * MN; __bf16* vb  = pbase + 5 * MN;
    __bf16* tta = pbase + 6 * MN; __bf16* ttb = pbase + 7 * MN;
    __bf16* ttbj = pbase + 8 * MN;

    float* out_a = (float*)d_out;
    float* out_b = out_a + MN;

    const dim3 cblk(256);
    const dim3 cgrdAct(MN / (8 * 256));   // 2048
    const dim3 tgrd(256);                 // 64x64 tiles over 1024x1024

    // fp32 -> bf16 activations (+ j2-prescaled b-components).
    cvt_kernel<<<cgrdAct, cblk, 0, stream>>>(x_q_a, xqa);
    cvt_pair_kernel<<<cgrdAct, cblk, 0, stream>>>(x_q_b, xqb, xqbj, layer_theta);
    cvt_kernel<<<cgrdAct, cblk, 0, stream>>>(x_k_a, xka);
    cvt_pair_kernel<<<cgrdAct, cblk, 0, stream>>>(x_k_b, xkb, xkbj, layer_theta);
    cvt_kernel<<<cgrdAct, cblk, 0, stream>>>(x_v_a, xva);
    cvt_pair_kernel<<<cgrdAct, cblk, 0, stream>>>(x_v_b, xvb, xvbj, layer_theta);
    // fp32 W[n][k] -> bf16 W^T[k][n].
    cvt_transpose_kernel<<<tgrd, cblk, 0, stream>>>(q_wa, qwa);
    cvt_transpose_kernel<<<tgrd, cblk, 0, stream>>>(q_wb, qwb);
    cvt_transpose_kernel<<<tgrd, cblk, 0, stream>>>(k_wa, kwa);
    cvt_transpose_kernel<<<tgrd, cblk, 0, stream>>>(k_wb, kwb);
    cvt_transpose_kernel<<<tgrd, cblk, 0, stream>>>(v_wa, vwa);
    cvt_transpose_kernel<<<tgrd, cblk, 0, stream>>>(v_wb, vwb);
    cvt_transpose_kernel<<<tgrd, cblk, 0, stream>>>(o_wa, owa);
    cvt_transpose_kernel<<<tgrd, cblk, 0, stream>>>(o_wb, owb);

    const dim3 blk(128);
    const dim3 projGrd(64 * 16);               // 1024 blocks, 4 waves each
    const dim3 attnGrd(2 * NH * (SEQ / 64));   // 1024 blocks

    qc_proj_bf16<0><<<projGrd, blk, 0, stream>>>(xqa, xqb, xqbj, qwa, qwb,
                                                 q_ba, q_bb, nullptr, nullptr,
                                                 qa, qb);
    qc_proj_bf16<1><<<projGrd, blk, 0, stream>>>(xka, xkb, xkbj, kwa, kwb,
                                                 k_ba, k_bb, nullptr, nullptr,
                                                 kaT, kbT);
    qc_proj_bf16<0><<<projGrd, blk, 0, stream>>>(xva, xvb, xvbj, vwa, vwb,
                                                 v_ba, v_bb, nullptr, nullptr,
                                                 va, vb);
    qc_attn_bf16<<<attnGrd, blk, 0, stream>>>(qa, qb, kaT, kbT, va, vb, mask,
                                              head_thetas, layer_theta,
                                              tta, ttb, ttbj);
    qc_proj_bf16<2><<<projGrd, blk, 0, stream>>>(tta, ttb, ttbj, owa, owb,
                                                 o_ba, o_bb, out_a, out_b,
                                                 nullptr, nullptr);
}
